// HGSN_11269994185378
// MI455X (gfx1250) — matrix operand order repaired and checked by reading, not compiled, not run on hardware
//
#include <hip/hip_runtime.h>
#include <hip/hip_bf16.h>

typedef __attribute__((ext_vector_type(16))) _Float16 v16h;
typedef __attribute__((ext_vector_type(8)))  float    v8f;

#define N1PAD 1056           // 1025 padded to multiple of 32
#define MROWS (8 * N1PAD)    // 8448
#define MEMB  (8 * 1024)     // 8192
#define DMODEL 512
#define ATT_SCALE 0.125f     // 64^-0.5

// ---------------------------------------------------------------------------
// WMMA helpers (contiguous k order, k = 16*(lane>>4) + e, same in A and B)
// ---------------------------------------------------------------------------
__device__ inline v8f wmma_f16(v16h a, v16h b, v8f c) {
  return __builtin_amdgcn_wmma_f32_16x16x32_f16(false, a, false, b, (short)0, c,
                                                false, false);
}

// A fragment: 16x32 f16, row-major source, row = lane&15.
// lanes 0-15: K 0..15 ; lanes 16-31: K 16..31
__device__ inline v16h load_a(const _Float16* base, int ld) {
  int lane = threadIdx.x & 31;
  const _Float16* p = base + (size_t)(lane & 15) * ld + ((lane >> 4) << 4);
  v16h f;
#pragma unroll
  for (int e = 0; e < 16; ++e) f[e] = p[e];
  return f;
}

// B fragment: 32x16 f16 from TRANSPOSED storage Bt[N][K] (row-major).
// column n = lane&15 ; lanes 0-15: K=e, lanes 16-31: K=16+e  -> 16 contiguous
__device__ inline v16h load_b(const _Float16* baseT, int ld) {
  int lane = threadIdx.x & 31;
  const _Float16* p = baseT + (size_t)(lane & 15) * ld + ((lane >> 4) << 4);
  v16h f;
#pragma unroll
  for (int e = 0; e < 16; ++e) f[e] = p[e];
  return f;
}

// B fragment directly from row-major B[K][N] (strided; used for V tiles)
__device__ inline v16h load_b_strided(const _Float16* base, int ld) {
  int lane = threadIdx.x & 31;
  const _Float16* p = base + (size_t)((lane >> 4) << 4) * ld + (lane & 15);
  v16h f;
#pragma unroll
  for (int e = 0; e < 16; ++e) f[e] = p[(size_t)e * ld];
  return f;
}

// ---------------------------------------------------------------------------
// Weight cast + transpose: src f32 [K][N] -> dst f16 [N][K]
// ---------------------------------------------------------------------------
__global__ void cast_transpose_kernel(const float* __restrict__ src,
                                      _Float16* __restrict__ dst, int K, int N) {
  int idx = blockIdx.x * blockDim.x + threadIdx.x;
  if (idx >= K * N) return;
  int n = idx / K, k = idx - n * K;
  dst[idx] = (_Float16)src[(size_t)k * N + n];
}

// ---------------------------------------------------------------------------
// Embedder gather: emb[b*1024+n][c] = c<512 ? te[type][c] : de[depth][c-512]
// ---------------------------------------------------------------------------
__global__ void embed_kernel(const int* __restrict__ types,
                             const int* __restrict__ depths,
                             const float* __restrict__ te,
                             const float* __restrict__ de,
                             _Float16* __restrict__ emb) {
  int idx = blockIdx.x * blockDim.x + threadIdx.x;
  if (idx >= MEMB * 1024) return;
  int row = idx >> 10, c = idx & 1023;
  float v;
  if (c < 512) {
    v = te[(size_t)types[row] * 512 + c];
  } else {
    int d = depths[row]; d = d < 0 ? 0 : (d > 63 ? 63 : d);
    v = de[(size_t)d * 512 + (c - 512)];
  }
  emb[idx] = (_Float16)v;
}

// ---------------------------------------------------------------------------
// Wave-per-macro-tile WMMA GEMM, 32x64 output per wave:
//   2 A fragments x 4 B fragments -> 8 accumulators, 8 WMMAs per K-step
//   (0.75 fragment loads per WMMA vs 2.0 for a naive 16x16 tile)
// mode 0: store f16   1: f16 + exact GELU   2: store f32   3: f32 residual +=
// ---------------------------------------------------------------------------
__global__ __launch_bounds__(32) void gemm_kernel(
    const _Float16* __restrict__ A, int lda,
    const _Float16* __restrict__ Bt, int ldb,
    const float* __restrict__ bias,
    void* __restrict__ C, int ldc, int K, int mode) {
  size_t arow = (size_t)blockIdx.x * 32;
  size_t bcol = (size_t)blockIdx.y * 64;
  const _Float16* Ab0 = A + arow * lda;
  const _Float16* Ab1 = A + (arow + 16) * lda;
  v8f c[2][4] = {};
  for (int k0 = 0; k0 < K; k0 += 32) {
    v16h a0 = load_a(Ab0 + k0, lda);
    v16h a1 = load_a(Ab1 + k0, lda);
#pragma unroll
    for (int j = 0; j < 4; ++j) {
      v16h b = load_b(Bt + (bcol + 16 * j) * ldb + k0, ldb);
      c[0][j] = wmma_f16(a0, b, c[0][j]);
      c[1][j] = wmma_f16(a1, b, c[1][j]);
    }
  }
  int lane = threadIdx.x & 31, nl = lane & 15, hi = lane >> 4;
#pragma unroll
  for (int j = 0; j < 4; ++j) {
    float bv = bias[bcol + 16 * j + nl];
#pragma unroll
    for (int i = 0; i < 2; ++i) {
#pragma unroll
      for (int r = 0; r < 8; ++r) {
        size_t row = arow + 16 * i + r + 8 * hi;
        float v = c[i][j][r] + bv;
        size_t idx = row * (size_t)ldc + bcol + 16 * j + nl;
        if (mode == 0) {
          ((_Float16*)C)[idx] = (_Float16)v;
        } else if (mode == 1) {
          ((_Float16*)C)[idx] =
              (_Float16)(0.5f * v * (1.0f + erff(v * 0.70710678f)));
        } else if (mode == 2) {
          ((float*)C)[idx] = v;
        } else {
          ((float*)C)[idx] += v;
        }
      }
    }
  }
}

// ---------------------------------------------------------------------------
// LayerNorm kernels (one row of 512 per block, 256 threads)
// ---------------------------------------------------------------------------
__device__ inline void ln_row_core(const float* p, const float* g,
                                   const float* bta, float* red, int t,
                                   float& o0, float& o1) {
  float v0 = p[t], v1 = p[t + 256];
  red[t] = v0 + v1;
  __syncthreads();
  for (int o = 128; o > 0; o >>= 1) { if (t < o) red[t] += red[t + o]; __syncthreads(); }
  float mean = red[0] * (1.0f / 512.0f);
  __syncthreads();
  float d0 = v0 - mean, d1 = v1 - mean;
  red[t] = d0 * d0 + d1 * d1;
  __syncthreads();
  for (int o = 128; o > 0; o >>= 1) { if (t < o) red[t] += red[t + o]; __syncthreads(); }
  float rstd = rsqrtf(red[0] * (1.0f / 512.0f) + 1e-5f);
  o0 = d0 * rstd * g[t] + bta[t];
  o1 = d1 * rstd * g[t + 256] + bta[t + 256];
}

__global__ void ln_f16_kernel(const float* __restrict__ in,
                              const float* __restrict__ g,
                              const float* __restrict__ bta,
                              _Float16* __restrict__ out) {
  __shared__ float red[256];
  int row = blockIdx.x, t = threadIdx.x;
  float o0, o1;
  ln_row_core(in + (size_t)row * 512, g, bta, red, t, o0, o1);
  out[(size_t)row * 512 + t] = (_Float16)o0;
  out[(size_t)row * 512 + t + 256] = (_Float16)o1;
}

// embedder LN: tmp row (b*1024+n) -> x row (b*1056 + 1 + n), f32
__global__ void ln_embed_kernel(const float* __restrict__ in,
                                const float* __restrict__ g,
                                const float* __restrict__ bta,
                                float* __restrict__ x) {
  __shared__ float red[256];
  int row = blockIdx.x, t = threadIdx.x;
  int b = row >> 10, n = row & 1023;
  float o0, o1;
  ln_row_core(in + (size_t)row * 512, g, bta, red, t, o0, o1);
  float* orow = x + (size_t)(b * N1PAD + 1 + n) * 512;
  orow[t] = o0; orow[t + 256] = o1;
}

// final LN on CLS row -> cls buffer and d_out[16 + b*512 ...]
__global__ void ln_final_kernel(const float* __restrict__ x,
                                const float* __restrict__ g,
                                const float* __restrict__ bta,
                                float* __restrict__ cls,
                                float* __restrict__ dout) {
  __shared__ float red[256];
  int b = blockIdx.x, t = threadIdx.x;
  float o0, o1;
  ln_row_core(x + (size_t)b * N1PAD * 512, g, bta, red, t, o0, o1);
  cls[(size_t)b * 512 + t] = o0;       cls[(size_t)b * 512 + t + 256] = o1;
  dout[16 + (size_t)b * 512 + t] = o0; dout[16 + (size_t)b * 512 + t + 256] = o1;
}

// fill CLS row with cls_tok and zero padding rows 1025..1055
__global__ void init_x_kernel(float* __restrict__ x,
                              const float* __restrict__ cls_tok) {
  int b = blockIdx.x, t = threadIdx.x;
  float* xb = x + (size_t)b * N1PAD * 512;
  xb[t] = cls_tok[t];
  xb[t + 256] = cls_tok[t + 256];
  for (int i = t; i < (N1PAD - 1025) * 512; i += 256) xb[1025 * 512 + i] = 0.0f;
}

// ---------------------------------------------------------------------------
// Fused flash attention with tree-distance bias.
// One wave per (16-query tile, head, batch). K-step = 32.
// ---------------------------------------------------------------------------
__device__ inline float sbval(float sraw, int q, int k, int b, int h,
                              const int* __restrict__ tree,
                              const unsigned char* __restrict__ pm,
                              const float* __restrict__ sb) {
  if (k >= 1025) return -1e30f;
  if (k > 0 && pm[b * 1024 + (k - 1)]) return -1e30f;
  int d;
  if (k == 0)          d = (q == 0) ? 0 : 1;
  else if (q == 0)     d = 1;
  else if (q >= 1025)  d = 1;  // padded query rows (discarded)
  else {
    d = tree[((size_t)b * 1024 + (q - 1)) * 1024 + (k - 1)];
    d = d < 0 ? 0 : (d > 31 ? 31 : d);
  }
  return sraw * ATT_SCALE + sb[d * 8 + h];
}

__global__ __launch_bounds__(32) void attn_kernel(
    const _Float16* __restrict__ qkv,      // [B*N1PAD][1536]
    const int* __restrict__ tree,          // [B][1024][1024]
    const unsigned char* __restrict__ pmask,
    const float* __restrict__ bias_l,      // bias_emb[layer]: [32][8]
    _Float16* __restrict__ o) {            // [B*N1PAD][512]
  __shared__ float sbias[32 * 8];
  __shared__ _Float16 pbuf[16 * 32];
  int lane = threadIdx.x & 31;
  int qt = blockIdx.x, h = blockIdx.y, b = blockIdx.z;
  for (int i = lane; i < 256; i += 32) sbias[i] = bias_l[i];
  __syncthreads();

  size_t brow = (size_t)b * N1PAD;
  int q0 = qt * 16, nl = lane & 15, hi = lane >> 4;

  const _Float16* Qb = qkv + (brow + q0) * 1536 + h * 64;
  v16h qa0 = load_a(Qb, 1536);
  v16h qa1 = load_a(Qb + 32, 1536);

  v8f acc[4] = {};
  float mrow[8], lrow[8];
#pragma unroll
  for (int r = 0; r < 8; ++r) { mrow[r] = -1e30f; lrow[r] = 0.0f; }

  for (int kt = 0; kt < N1PAD; kt += 32) {
    const _Float16* Kb = qkv + (brow + kt) * 1536 + 512 + h * 64;
    v16h kb00 = load_b(Kb, 1536);                // keys kt..kt+15, d 0..31
    v16h kb01 = load_b(Kb + 32, 1536);           // d 32..63
    v16h kb10 = load_b(Kb + 16 * 1536, 1536);    // keys kt+16..kt+31
    v16h kb11 = load_b(Kb + 16 * 1536 + 32, 1536);
    v8f s0 = {}, s1 = {};
    s0 = wmma_f16(qa0, kb00, s0);  s0 = wmma_f16(qa1, kb01, s0);
    s1 = wmma_f16(qa0, kb10, s1);  s1 = wmma_f16(qa1, kb11, s1);

    float p0[8], p1[8];
#pragma unroll
    for (int r = 0; r < 8; ++r) {
      int q = q0 + r + 8 * hi;
      float e0 = sbval(s0[r], q, kt + nl, b, h, tree, pmask, sbias);
      float e1 = sbval(s1[r], q, kt + 16 + nl, b, h, tree, pmask, sbias);
      float mx = fmaxf(e0, e1);
#pragma unroll
      for (int off = 8; off > 0; off >>= 1) mx = fmaxf(mx, __shfl_xor(mx, off, 32));
      float mnew = fmaxf(mrow[r], mx);
      float pe0 = __expf(e0 - mnew), pe1 = __expf(e1 - mnew);
      float ps = pe0 + pe1;
#pragma unroll
      for (int off = 8; off > 0; off >>= 1) ps += __shfl_xor(ps, off, 32);
      float corr = __expf(mrow[r] - mnew);
      lrow[r] = lrow[r] * corr + ps;
      mrow[r] = mnew;
#pragma unroll
      for (int j = 0; j < 4; ++j) acc[j][r] *= corr;
      p0[r] = pe0; p1[r] = pe1;
    }

    __syncthreads();   // protect pbuf vs previous iteration's reads
#pragma unroll
    for (int r = 0; r < 8; ++r) {
      pbuf[(r + 8 * hi) * 32 + nl] = (_Float16)p0[r];
      pbuf[(r + 8 * hi) * 32 + 16 + nl] = (_Float16)p1[r];
    }
    __syncthreads();

    v16h pa = load_a((const _Float16*)pbuf, 32);
    const _Float16* Vb = qkv + (brow + kt) * 1536 + 1024 + h * 64;
#pragma unroll
    for (int j = 0; j < 4; ++j) {
      v16h vb = load_b_strided(Vb + j * 16, 1536);
      acc[j] = wmma_f16(pa, vb, acc[j]);
    }
  }

#pragma unroll
  for (int j = 0; j < 4; ++j)
#pragma unroll
    for (int r = 0; r < 8; ++r) {
      size_t row = brow + q0 + r + 8 * hi;
      float val = acc[j][r] / lrow[r];
      o[row * 512 + h * 64 + j * 16 + nl] = (_Float16)val;
    }
}

// ---------------------------------------------------------------------------
// Heads: detect = gelu(cls@W1+b1)@W2+b2 ; sev = sigmoid(relu(cls@S1+s1)@S2+s2)
// ---------------------------------------------------------------------------
__global__ void heads_kernel(const float* __restrict__ cls,
                             const float* __restrict__ d1w, const float* __restrict__ d1b,
                             const float* __restrict__ d2w, const float* __restrict__ d2b,
                             const float* __restrict__ s1w, const float* __restrict__ s1b,
                             const float* __restrict__ s2w, const float* __restrict__ s2b,
                             float* __restrict__ dout) {
  __shared__ float red[256];
  int b = blockIdx.x, t = threadIdx.x;
  const float* c = cls + (size_t)b * 512;
  float a = d1b[t];
  for (int d = 0; d < 512; ++d) a += c[d] * d1w[(size_t)d * 256 + t];
  float g = 0.5f * a * (1.0f + erff(a * 0.70710678f));
  red[t] = g * d2w[t];
  __syncthreads();
  for (int o = 128; o > 0; o >>= 1) { if (t < o) red[t] += red[t + o]; __syncthreads(); }
  if (t == 0) dout[b] = red[0] + d2b[0];
  __syncthreads();
  float sv = 0.0f;
  if (t < 32) {
    float a2 = s1b[t];
    for (int d = 0; d < 512; ++d) a2 += c[d] * s1w[(size_t)d * 32 + t];
    a2 = fmaxf(a2, 0.0f);
    sv = a2 * s2w[t];
  }
  red[t] = sv;
  __syncthreads();
  for (int o = 128; o > 0; o >>= 1) { if (t < o) red[t] += red[t + o]; __syncthreads(); }
  if (t == 0) dout[8 + b] = 1.0f / (1.0f + __expf(-(red[0] + s2b[0])));
}

// ---------------------------------------------------------------------------
// Host orchestration
// ---------------------------------------------------------------------------
extern "C" void kernel_launch(void* const* d_in, const int* in_sizes, int n_in,
                              void* d_out, int out_size, void* d_ws, size_t ws_size,
                              hipStream_t stream) {
  (void)in_sizes; (void)n_in; (void)out_size; (void)ws_size;
  const int*   node_type = (const int*)d_in[0];
  const int*   depths    = (const int*)d_in[1];
  const int*   tree      = (const int*)d_in[2];
  const unsigned char* pmask = (const unsigned char*)d_in[3];
  const float* te     = (const float*)d_in[4];
  const float* de     = (const float*)d_in[5];
  const float* proj_w = (const float*)d_in[6];
  const float* proj_b = (const float*)d_in[7];
  const float* lne_g  = (const float*)d_in[8];
  const float* lne_b  = (const float*)d_in[9];
  const float* cls_tok= (const float*)d_in[10];
  const float* qkv_w  = (const float*)d_in[11];
  const float* qkv_b  = (const float*)d_in[12];
  const float* out_w  = (const float*)d_in[13];
  const float* out_b  = (const float*)d_in[14];
  const float* bias_emb = (const float*)d_in[15];
  const float* n1_g = (const float*)d_in[16];
  const float* n1_b = (const float*)d_in[17];
  const float* n2_g = (const float*)d_in[18];
  const float* n2_b = (const float*)d_in[19];
  const float* ff1_w = (const float*)d_in[20];
  const float* ff1_b = (const float*)d_in[21];
  const float* ff2_w = (const float*)d_in[22];
  const float* ff2_b = (const float*)d_in[23];
  const float* fn_g = (const float*)d_in[24];
  const float* fn_b = (const float*)d_in[25];
  const float* det1_w = (const float*)d_in[26];
  const float* det1_b = (const float*)d_in[27];
  const float* det2_w = (const float*)d_in[28];
  const float* det2_b = (const float*)d_in[29];
  const float* sev1_w = (const float*)d_in[30];
  const float* sev1_b = (const float*)d_in[31];
  const float* sev2_w = (const float*)d_in[32];
  const float* sev2_b = (const float*)d_in[33];
  float* dout = (float*)d_out;

  // workspace carve-up
  char* wp = (char*)d_ws;
  auto alloc = [&](size_t bytes) {
    void* p = (void*)wp;
    wp += (bytes + 255) & ~(size_t)255;
    return p;
  };
  float*     x    = (float*)alloc((size_t)MROWS * 512 * 4);
  _Float16*  hbuf = (_Float16*)alloc((size_t)MROWS * 512 * 2);
  _Float16*  qkvb = (_Float16*)alloc((size_t)MROWS * 1536 * 2);
  _Float16*  obuf = (_Float16*)alloc((size_t)MROWS * 512 * 2);
  _Float16*  ffb  = (_Float16*)alloc((size_t)MROWS * 2048 * 2);
  float*     tmp  = (float*)alloc((size_t)MEMB * 512 * 4);
  _Float16*  emb  = (_Float16*)alloc((size_t)MEMB * 1024 * 2);
  _Float16*  projT= (_Float16*)alloc((size_t)512 * 1024 * 2);
  _Float16*  qkvT = (_Float16*)alloc((size_t)6 * 1536 * 512 * 2);
  _Float16*  outT = (_Float16*)alloc((size_t)6 * 512 * 512 * 2);
  _Float16*  ff1T = (_Float16*)alloc((size_t)6 * 2048 * 512 * 2);
  _Float16*  ff2T = (_Float16*)alloc((size_t)6 * 512 * 2048 * 2);
  float*     clsb = (float*)alloc((size_t)8 * 512 * 4);

  auto ct = [&](const float* src, _Float16* dst, int K, int N) {
    int total = K * N;
    cast_transpose_kernel<<<(total + 255) / 256, 256, 0, stream>>>(src, dst, K, N);
  };
  auto gemm = [&](const _Float16* A, int lda, const _Float16* Bt, int ldb,
                  const float* bias, void* C, int ldc, int M, int N, int K, int mode) {
    dim3 g(M / 32, N / 64);
    gemm_kernel<<<g, 32, 0, stream>>>(A, lda, Bt, ldb, bias, C, ldc, K, mode);
  };

  // ---- weight precision/layout conversion (f32 [K][N] -> f16 [N][K]) ----
  ct(proj_w, projT, 1024, 512);
  for (int l = 0; l < 6; ++l) {
    ct(qkv_w + (size_t)l * 512 * 1536, qkvT + (size_t)l * 1536 * 512, 512, 1536);
    ct(out_w + (size_t)l * 512 * 512,  outT + (size_t)l * 512 * 512,  512, 512);
    ct(ff1_w + (size_t)l * 512 * 2048, ff1T + (size_t)l * 2048 * 512, 512, 2048);
    ct(ff2_w + (size_t)l * 2048 * 512, ff2T + (size_t)l * 512 * 2048, 2048, 512);
  }

  // ---- embedder ----
  embed_kernel<<<(MEMB * 1024 + 255) / 256, 256, 0, stream>>>(node_type, depths, te, de, emb);
  gemm(emb, 1024, projT, 1024, proj_b, tmp, 512, MEMB, 512, 1024, /*f32*/2);
  init_x_kernel<<<8, 256, 0, stream>>>(x, cls_tok);
  ln_embed_kernel<<<MEMB, 256, 0, stream>>>(tmp, lne_g, lne_b, x);

  // ---- transformer layers ----
  for (int l = 0; l < 6; ++l) {
    ln_f16_kernel<<<MROWS, 256, 0, stream>>>(x, n1_g + l * 512, n1_b + l * 512, hbuf);
    gemm(hbuf, 512, qkvT + (size_t)l * 1536 * 512, 512, qkv_b + (size_t)l * 1536,
         qkvb, 1536, MROWS, 1536, 512, /*f16*/0);
    dim3 ag(N1PAD / 16, 8, 8);
    attn_kernel<<<ag, 32, 0, stream>>>(qkvb, tree, pmask,
                                       bias_emb + (size_t)l * 32 * 8, obuf);
    gemm(obuf, 512, outT + (size_t)l * 512 * 512, 512, out_b + (size_t)l * 512,
         x, 512, MROWS, 512, 512, /*res*/3);
    ln_f16_kernel<<<MROWS, 256, 0, stream>>>(x, n2_g + l * 512, n2_b + l * 512, hbuf);
    gemm(hbuf, 512, ff1T + (size_t)l * 2048 * 512, 512, ff1_b + (size_t)l * 2048,
         ffb, 2048, MROWS, 2048, 512, /*gelu f16*/1);
    gemm(ffb, 2048, ff2T + (size_t)l * 512 * 2048, 2048, ff2_b + (size_t)l * 512,
         x, 512, MROWS, 512, 2048, /*res*/3);
  }

  // ---- heads ----
  ln_final_kernel<<<8, 256, 0, stream>>>(x, fn_g, fn_b, clsb, dout);
  heads_kernel<<<8, 256, 0, stream>>>(clsb, det1_w, det1_b, det2_w, det2_b,
                                      sev1_w, sev1_b, sev2_w, sev2_b, dout);
}
